// PureSparseBackboneGeo_Res1_38079180046733
// MI455X (gfx1250) — matrix operand order repaired and checked by reading, not compiled, not run on hardware
//
#include <hip/hip_runtime.h>
#include <hip/hip_bf16.h>
#include <stdint.h>

// ---------------------------------------------------------------------------
// MI455X (gfx1250) sparse-conv ResNet backbone.
// Heavy convs use V_WMMA_F32_16X16X32_BF16 (bf16 A/B, f32 accumulate).
// Weights are pre-packed into WMMA B-fragment layout and staged via LDS
// with GLOBAL_LOAD_ASYNC_TO_LDS_B128 (ASYNCcnt path).
// Activations / BN statistics / residuals stay in f32.
// ---------------------------------------------------------------------------

typedef __attribute__((ext_vector_type(16))) __bf16 v16bf;
typedef __attribute__((ext_vector_type(8)))  __bf16 v8bf;
typedef __attribute__((ext_vector_type(8)))  float  v8f;
typedef __attribute__((ext_vector_type(4)))  int    v4i;

#define GAS __attribute__((address_space(1)))   // global
#define LAS __attribute__((address_space(3)))   // LDS

#if defined(__has_builtin)
# if __has_builtin(__builtin_amdgcn_global_load_async_to_lds_b128) && \
     __has_builtin(__builtin_amdgcn_s_wait_asynccnt)
#  define ATHENA_ASYNC_LDS 1
# endif
#endif
#ifndef ATHENA_ASYNC_LDS
# define ATHENA_ASYNC_LDS 0
#endif

static constexpr int kN0 = 200000;
static constexpr int kN1 = 160000;
static constexpr int kN2 = 100000;
static constexpr int kRED = 240;          // blocks used by BN reduction
static constexpr float kEps = 1e-5f;

#define DI __device__ __forceinline__

DI __bf16 f2bf(float f) {
  unsigned u = __builtin_bit_cast(unsigned, f);
  unsigned r = (u + 0x7FFFu + ((u >> 16) & 1u)) >> 16;   // round-nearest-even
  unsigned short h = (unsigned short)r;
  return __builtin_bit_cast(__bf16, h);
}

// ---------------------------------------------------------------------------
// Pack f32 weights (KT, CIN, COUT) into bf16 WMMA B-fragments.
// Fragment (k, kc, nt) = 512 bf16: lane l<16 -> N=nt*16+l, K=kc*32+j (j=0..15)
//                        lane l>=16 -> N=nt*16+l-16, K=kc*32+16+j
// ---------------------------------------------------------------------------
template<int CIN, int COUT, int KT>
__global__ void pack_w_kernel(const float* __restrict__ W, __bf16* __restrict__ packw) {
  constexpr int KC = CIN / 32, NT = COUT / 16;
  const int total = KT * KC * NT * 512;
  int i = blockIdx.x * 256 + threadIdx.x;
  if (i >= total) return;
  int j    = i & 15;
  int lane = (i >> 4) & 31;
  int frag = i >> 9;
  int nt = frag % NT;
  int kc = (frag / NT) % KC;
  int k  = frag / (NT * KC);
  int n   = nt * 16 + (lane & 15);
  int kin = kc * 32 + ((lane >> 4) ? 16 : 0) + j;
  packw[i] = f2bf(W[((long)k * CIN + kin) * COUT + n]);
}

// ---------------------------------------------------------------------------
// WMMA gather-conv:  out[n, :] = sum_k  feat[nbr[k][n], :] @ Wk
// One wave per 16-row tile; B staged through LDS in <=60KB chunks.
// feat has (Nin+1) rows; row Nin is all-zero (padding target).
// ---------------------------------------------------------------------------
template<int CIN, int COUT, int KT>
__global__ __launch_bounds__(256)
void conv_wmma_kernel(const __bf16* __restrict__ feat,
                      const int*    __restrict__ nbr,
                      const __bf16* __restrict__ packw,
                      float*        __restrict__ out,
                      int nrows) {
  constexpr int KC   = CIN / 32;
  constexpr int NT   = COUT / 16;
  constexpr int PERK = KC * NT * 512;                       // bf16 elems per tap
  constexpr int KCH  = (KT * PERK * 2 <= 61440) ? KT : (61440 / (PERK * 2));
  __shared__ __bf16 lw[(size_t)KCH * PERK];

  const int tid  = threadIdx.x;
  const int lane = tid & 31;
  const int wave = tid >> 5;
  const int m    = lane & 15;
  const int half = lane >> 4;
  const long n0  = ((long)blockIdx.x * 8 + wave) * 16;
  const bool active = (n0 < nrows);

  v8f c[NT] = {};

  for (int k0 = 0; k0 < KT; k0 += KCH) {
    const int kn = (KT - k0 < KCH) ? (KT - k0) : KCH;
    __syncthreads();                                        // protect prior readers
#if ATHENA_ASYNC_LDS
    {
      // Bulk weight staging through the CDNA5 async path: data flows
      // global -> LDS without touching VGPRs; tracked by ASYNCcnt.
      v4i* gg = (v4i*)(void*)const_cast<__bf16*>(packw + (long)k0 * PERK);
      GAS v4i* g = (GAS v4i*)gg;
      LAS v4i* l = (LAS v4i*)(v4i*)(void*)lw;
      const int nvec = (kn * PERK) / 8;                     // 16B units
      for (int i = tid; i < nvec; i += 256)
        __builtin_amdgcn_global_load_async_to_lds_b128(g + i, l + i, 0, 0);
      __builtin_amdgcn_s_wait_asynccnt(0);
    }
#else
    for (int i = tid * 8; i < kn * PERK; i += 256 * 8)
      *(v8bf*)(lw + i) = *(const v8bf*)(packw + (long)k0 * PERK + i);
#endif
    __syncthreads();
    if (!active) continue;

    for (int k = 0; k < kn; k++) {
      const int idx = nbr[(long)(k0 + k) * nrows + n0 + m];
      const __bf16* row = feat + (long)idx * CIN;
#pragma unroll
      for (int kc = 0; kc < KC; kc++) {
        // A fragment, contiguous k order (same as packed B):
        //  lanes 0-15 : K = kc*32 + [0..15]
        //  lanes 16-31: K = kc*32 + [16..31]
        union { v16bf v; v8bf h[2]; } a;
        a.h[0] = *(const v8bf*)(row + kc * 32 + half * 16);
        a.h[1] = *(const v8bf*)(row + kc * 32 + half * 16 + 8);
        const __bf16* lb = lw + (size_t)(k * KC + kc) * NT * 512 + lane * 16;
#pragma unroll
        for (int nt = 0; nt < NT; nt++) {
          v16bf b = *(const v16bf*)(lb + nt * 512);
          c[nt] = __builtin_amdgcn_wmma_f32_16x16x32_bf16(
              false, a.v, false, b, (short)0, c[nt], false, false);
        }
      }
    }
  }
  if (!active) return;

  // C/D layout: lane<16 -> N=lane, VGPR r = M=r ; lane>=16 -> M=8+r
#pragma unroll
  for (int nt = 0; nt < NT; nt++) {
#pragma unroll
    for (int r = 0; r < 8; r++) {
      const long rowi = n0 + r + half * 8;
      out[rowi * COUT + nt * 16 + m] = c[nt][r];
    }
  }
}

// ---------------------------------------------------------------------------
// conv1: C_in = 1, 125 taps, fused ReLU + bf16 emit. One thread per row.
// ---------------------------------------------------------------------------
__global__ __launch_bounds__(256)
void conv1_kernel(const float* __restrict__ feats, const int* __restrict__ nbr1,
                  const float* __restrict__ W1, float* __restrict__ x1,
                  __bf16* __restrict__ xb1) {
  __shared__ float w[125 * 32];
  for (int i = threadIdx.x; i < 125 * 32; i += 256) w[i] = W1[i];
  __syncthreads();
  const long n = (long)blockIdx.x * 256 + threadIdx.x;
  if (n >= kN1) return;
  float acc[32];
#pragma unroll
  for (int f = 0; f < 32; f++) acc[f] = 0.f;
  for (int k = 0; k < 125; k++) {
    const int idx = nbr1[(long)k * kN1 + n];
    const float v = (idx < kN0) ? feats[idx] : 0.f;
#pragma unroll
    for (int f = 0; f < 32; f++) acc[f] = fmaf(v, w[k * 32 + f], acc[f]);
  }
#pragma unroll
  for (int f = 0; f < 32; f++) {
    const float r = fmaxf(acc[f], 0.f);
    x1[n * 32 + f]  = r;
    xb1[n * 32 + f] = f2bf(r);
  }
}

// ---------------------------------------------------------------------------
// BatchNorm: per-block partial sum / sumsq  ->  finalize scale/shift
// ---------------------------------------------------------------------------
template<int C>
__global__ __launch_bounds__(256)
void bn_reduce_kernel(const float* __restrict__ t, int nrows,
                      float* __restrict__ partials) {
  constexpr int G = 256 / C;
  const int tid = threadIdx.x;
  const int ch  = tid & (C - 1);
  const int g   = tid / C;
  float s = 0.f, s2 = 0.f;
  for (long r = (long)blockIdx.x * G + g; r < nrows; r += (long)gridDim.x * G) {
    const float v = t[r * C + ch];
    s += v; s2 += v * v;
  }
  __shared__ float ls[256], ls2[256];
  ls[tid] = s; ls2[tid] = s2;
  __syncthreads();
  for (int st = 128; st >= C; st >>= 1) {
    if (tid < st) { ls[tid] += ls[tid + st]; ls2[tid] += ls2[tid + st]; }
    __syncthreads();
  }
  if (tid < C) {
    partials[(long)blockIdx.x * 2 * C + tid]     = ls[tid];
    partials[(long)blockIdx.x * 2 * C + C + tid] = ls2[tid];
  }
}

template<int C>
__global__ void bn_finalize_kernel(const float* __restrict__ partials, int nblk,
                                   const float* __restrict__ g,
                                   const float* __restrict__ b,
                                   long nrows, float* __restrict__ ss) {
  const int c = threadIdx.x;
  if (c >= C) return;
  float s = 0.f, s2 = 0.f;
  for (int i = 0; i < nblk; i++) {
    s  += partials[(long)i * 2 * C + c];
    s2 += partials[(long)i * 2 * C + C + c];
  }
  const float inv = 1.0f / (float)nrows;
  const float mu  = s * inv;
  const float var = s2 * inv - mu * mu;
  const float sc  = g[c] * rsqrtf(var + kEps);
  ss[c]     = sc;
  ss[C + c] = b[c] - mu * sc;
}

// y = relu(t*scale + shift)  -> bf16 (feeds next gather-conv)
template<int C>
__global__ void bn_apply_relu_kernel(const float* __restrict__ t,
                                     const float* __restrict__ ss,
                                     __bf16* __restrict__ yb, long nrows) {
  const long i = (long)blockIdx.x * 256 + threadIdx.x;
  if (i >= nrows * C) return;
  const int c = (int)(i & (C - 1));
  yb[i] = f2bf(fmaxf(t[i] * ss[c] + ss[C + c], 0.f));
}

// xnew = relu(x + t*scale + shift); optional bf16 emit
template<int C, bool WRITE_BF>
__global__ void bn_residual_kernel(const float* __restrict__ t,
                                   const float* __restrict__ ss,
                                   const float* __restrict__ x,
                                   float* __restrict__ xo,
                                   __bf16* __restrict__ xb, long nrows) {
  const long i = (long)blockIdx.x * 256 + threadIdx.x;
  if (i >= nrows * C) return;
  const int c = (int)(i & (C - 1));
  const float v = fmaxf(x[i] + t[i] * ss[c] + ss[C + c], 0.f);
  xo[i] = v;
  if (WRITE_BF) xb[i] = f2bf(v);
}

template<int C>
__global__ void to_bf16_kernel(const float* __restrict__ x, __bf16* __restrict__ xb,
                               long total) {
  const long i = (long)blockIdx.x * 256 + threadIdx.x;
  if (i < total) xb[i] = f2bf(x[i]);
}

__global__ void zero_pads_kernel(__bf16* xb1, __bf16* xb2) {
  const int t = threadIdx.x;
  if (t < 32) xb1[(long)kN1 * 32 + t] = f2bf(0.f);
  if (t < 64) xb2[(long)kN2 * 64 + t] = f2bf(0.f);
}

// ---------------------------------------------------------------------------
// Host-side orchestration helpers
// ---------------------------------------------------------------------------
template<int CIN, int COUT>
static void launch_conv(const __bf16* feat, const int* nbr, const float* W,
                        __bf16* packw, float* out, long nrows, hipStream_t s) {
  constexpr int KT = 27;
  constexpr int total = KT * (CIN / 32) * (COUT / 16) * 512;
  pack_w_kernel<CIN, COUT, KT><<<(total + 255) / 256, 256, 0, s>>>(W, packw);
  const long tiles = nrows / 16;
  conv_wmma_kernel<CIN, COUT, KT><<<(unsigned)((tiles + 7) / 8), 256, 0, s>>>(
      feat, nbr, packw, out, (int)nrows);
}

template<int C>
static void launch_bn_stats(const float* t, long nrows, float* partials,
                            const float* g, const float* b, float* ss,
                            hipStream_t s) {
  bn_reduce_kernel<C><<<kRED, 256, 0, s>>>(t, (int)nrows, partials);
  bn_finalize_kernel<C><<<1, C, 0, s>>>(partials, kRED, g, b, nrows, ss);
}

template<int C>
static void run_basic_block(const float* wa, const float* wb,
                            const float* g1, const float* b1,
                            const float* g2, const float* b2,
                            const int* nbr, long nrows,
                            float* x, __bf16* xb, float* t,
                            __bf16* packw, float* partials, float* ss,
                            float* final_out, hipStream_t s) {
  const long total = nrows * C;
  const unsigned eg = (unsigned)((total + 255) / 256);

  launch_conv<C, C>(xb, nbr, wa, packw, t, nrows, s);
  launch_bn_stats<C>(t, nrows, partials, g1, b1, ss, s);
  bn_apply_relu_kernel<C><<<eg, 256, 0, s>>>(t, ss, xb, nrows);   // xb := bf16(y)

  launch_conv<C, C>(xb, nbr, wb, packw, t, nrows, s);
  launch_bn_stats<C>(t, nrows, partials, g2, b2, ss, s);
  if (final_out)
    bn_residual_kernel<C, false><<<eg, 256, 0, s>>>(t, ss, x, final_out, nullptr, nrows);
  else
    bn_residual_kernel<C, true><<<eg, 256, 0, s>>>(t, ss, x, x, xb, nrows);
}

static constexpr size_t alignUp(size_t v) { return (v + 255) & ~(size_t)255; }

extern "C" void kernel_launch(void* const* d_in, const int* in_sizes, int n_in,
                              void* d_out, int out_size, void* d_ws, size_t ws_size,
                              hipStream_t stream) {
  (void)in_sizes; (void)n_in; (void)out_size; (void)ws_size;

  const float* feats  = (const float*)d_in[0];
  const float* W1     = (const float*)d_in[1];
  const float* W2     = (const float*)d_in[2];
  const int*   nbr1   = (const int*)d_in[3];
  const int*   nbr_e1 = (const int*)d_in[4];
  const int*   nbr2   = (const int*)d_in[5];
  const int*   nbr_e2 = (const int*)d_in[6];
  // blocks: e1b1, e1b2, e2b1, e2b2 -> wa, wb, g1, b1, g2, b2
  const float* blk[4][6];
  for (int i = 0; i < 4; i++)
    for (int j = 0; j < 6; j++)
      blk[i][j] = (const float*)d_in[7 + i * 6 + j];

  // workspace carve-up
  char* base = (char*)d_ws;
  size_t off = 0;
  auto take = [&](size_t bytes) { char* p = base + off; off += alignUp(bytes); return p; };
  float*  X1   = (float*) take((size_t)kN1 * 32 * 4);
  float*  T1   = (float*) take((size_t)kN1 * 32 * 4);
  __bf16* XB1  = (__bf16*)take((size_t)(kN1 + 1) * 32 * 2);
  float*  X2   = (float*) take((size_t)kN2 * 64 * 4);
  float*  T2   = (float*) take((size_t)kN2 * 64 * 4);
  __bf16* XB2  = (__bf16*)take((size_t)(kN2 + 1) * 64 * 2);
  __bf16* PACK = (__bf16*)take((size_t)27 * 2 * 4 * 512 * 2);   // max 64x64 pack
  float*  PART = (float*) take((size_t)kRED * 2 * 64 * 4);
  float*  SS   = (float*) take((size_t)2 * 64 * 4);

  zero_pads_kernel<<<1, 64, 0, stream>>>(XB1, XB2);

  // stage 1: conv1 (+ReLU) -> X1 / XB1
  conv1_kernel<<<(kN1 + 255) / 256, 256, 0, stream>>>(feats, nbr1, W1, X1, XB1);

  // stage 2: two 32-ch residual blocks on N1 rows
  run_basic_block<32>(blk[0][0], blk[0][1], blk[0][2], blk[0][3], blk[0][4], blk[0][5],
                      nbr_e1, kN1, X1, XB1, T1, PACK, PART, SS, nullptr, stream);
  run_basic_block<32>(blk[1][0], blk[1][1], blk[1][2], blk[1][3], blk[1][4], blk[1][5],
                      nbr_e1, kN1, X1, XB1, T1, PACK, PART, SS, nullptr, stream);

  // stage 3: downsample conv 32 -> 64 on N2 rows (no activation)
  launch_conv<32, 64>(XB1, nbr2, W2, PACK, X2, kN2, stream);
  to_bf16_kernel<64><<<(unsigned)(((long)kN2 * 64 + 255) / 256), 256, 0, stream>>>(
      X2, XB2, (long)kN2 * 64);

  // stage 4: two 64-ch residual blocks; final one writes d_out
  run_basic_block<64>(blk[2][0], blk[2][1], blk[2][2], blk[2][3], blk[2][4], blk[2][5],
                      nbr_e2, kN2, X2, XB2, T2, PACK, PART, SS, nullptr, stream);
  run_basic_block<64>(blk[3][0], blk[3][1], blk[3][2], blk[3][3], blk[3][4], blk[3][5],
                      nbr_e2, kN2, X2, XB2, T2, PACK, PART, SS, (float*)d_out, stream);
}
